// SubGraphTransformer_8435315769789
// MI455X (gfx1250) — compile-verified, matrix operand order checked by reading, not run on hardware
//
#include <hip/hip_runtime.h>
#include <hip/hip_bf16.h>

// ---------------------------------------------------------------------------
// SubGraphTransformer for MI455X (gfx1250, wave32, WMMA).
// All dense matmuls go through v_wmma_f32_16x16x32_f16 (f16 in, f32 acc).
// GEMM waves compute 16x64 tiles (4 WMMAs back-to-back, A reused x4).
// Fragments loaded with float4 (global_load_b128 / ds_load_b128).
// Floyd-Warshall stays VALU (min-plus) with global_prefetch of step k+1.
// ---------------------------------------------------------------------------

#define BB 8
#define NN 512
#define EE 256
#define HH 8
#define HD 32
#define FF 1024
#define TOPK 102

typedef __attribute__((ext_vector_type(16))) _Float16 v16h;
typedef __attribute__((ext_vector_type(8)))  float    v8f;

// ---- WMMA fragment loaders (CDNA5 16x16x32 f16 layouts, wave32) -----------
// A (16x32, MxK): lane L<16 holds row M=L, K = {0..7, 16..23}; lanes 16..31
// hold K = {8..15, 24..31}.  B (32x16, KxN) mirrors with N in lane[3:0].

// Source row-major, K contiguous (X[M][K] for A, W[N][K] for B=W^T).
// 16-byte aligned rows (ld multiple of 4, k0 multiple of 8) -> b128 loads.
__device__ __forceinline__ v16h load_frag_rowK(const float* base, int ld,
                                               int row0, int k0, int lane) {
  const float* p = base + (size_t)(row0 + (lane & 15)) * ld + k0 + ((lane >> 4) << 3);
  float4 x0 = ((const float4*)p)[0];
  float4 x1 = ((const float4*)p)[1];
  float4 y0 = ((const float4*)(p + 16))[0];
  float4 y1 = ((const float4*)(p + 16))[1];
  v16h f;
  f[0]  = (_Float16)x0.x; f[1]  = (_Float16)x0.y;
  f[2]  = (_Float16)x0.z; f[3]  = (_Float16)x0.w;
  f[4]  = (_Float16)x1.x; f[5]  = (_Float16)x1.y;
  f[6]  = (_Float16)x1.z; f[7]  = (_Float16)x1.w;
  f[8]  = (_Float16)y0.x; f[9]  = (_Float16)y0.y;
  f[10] = (_Float16)y0.z; f[11] = (_Float16)y0.w;
  f[12] = (_Float16)y1.x; f[13] = (_Float16)y1.y;
  f[14] = (_Float16)y1.z; f[15] = (_Float16)y1.w;
  return f;
}

// B stored K-major: B[k][n] = base[k*ld + n]  (e.g. V matrix [keys][d]).
__device__ __forceinline__ v16h load_frag_colK(const float* base, int ld,
                                               int k0, int n0, int lane) {
  const float* p = base + (size_t)(k0 + ((lane >> 4) << 3)) * ld + n0 + (lane & 15);
  v16h f;
#pragma unroll
  for (int i = 0; i < 8; ++i) f[i] = (_Float16)p[(size_t)i * ld];
#pragma unroll
  for (int i = 0; i < 8; ++i) f[8 + i] = (_Float16)p[(size_t)(16 + i) * ld];
  return f;
}

__device__ __forceinline__ v8f wmma_f16(v16h a, v16h b, v8f c) {
  return __builtin_amdgcn_wmma_f32_16x16x32_f16(false, a, false, b,
                                                (short)0, c, false, false);
}

__device__ __forceinline__ float wave_reduce_sum(float v) {
#pragma unroll
  for (int o = 16; o > 0; o >>= 1) v += __shfl_xor(v, o, 32);
  return v;
}

// ---------------------------------------------------------------------------
// 1) top-k per row -> binary adjacency + FW distance init
// ---------------------------------------------------------------------------
__global__ __launch_bounds__(512) void topk_binary_dist_kernel(
    const float* __restrict__ adj, unsigned char* __restrict__ binary,
    float* __restrict__ dist) {
  int row = blockIdx.x;             // b*N + i
  int i = row & (NN - 1);
  int t = threadIdx.x;              // column j
  __shared__ float s[NN];
  const float* a = adj + (size_t)row * NN;
  s[t] = a[t];
  __syncthreads();
  float v = s[t];
  int rank = 0;
#pragma unroll 8
  for (int l = 0; l < NN; ++l) {
    float w = s[l];
    rank += (w > v) || (w == v && l < t);   // stable top-k tie-break
  }
  int isTop = rank < TOPK;
  binary[(size_t)row * NN + t] = (unsigned char)isTop;
  float d = isTop ? 1.0f : __builtin_inff();
  if (t == i) d = 0.0f;                     // diagonal forced 0 after init
  dist[(size_t)row * NN + t] = d;
}

// ---------------------------------------------------------------------------
// 2) Floyd-Warshall: one workgroup per batch, row-k/col-k staged in LDS.
//    Row k and column k are fixed points of iteration k -> race-free.
//    Prefetch step k+1's row/col while relaxing step k.
// ---------------------------------------------------------------------------
__global__ __launch_bounds__(1024) void floyd_warshall_kernel(float* __restrict__ dist) {
  float* d = dist + (size_t)blockIdx.x * NN * NN;
  __shared__ float rowk[NN];
  __shared__ float colk[NN];
  int t = threadIdx.x;              // 0..1023
  int i = t >> 1;                   // owned row
  int j0 = (t & 1) << 8;            // owned half-row (256 cols)
  for (int k = 0; k < NN; ++k) {
    if (t < NN) rowk[t] = d[(size_t)k * NN + t];
    else        colk[t - NN] = d[(size_t)(t - NN) * NN + k];
    if (k + 1 < NN) {               // overlap next staging with this relax
      if (t < NN) __builtin_prefetch(&d[(size_t)(k + 1) * NN + t], 0, 0);
      else        __builtin_prefetch(&d[(size_t)(t - NN) * NN + (k + 1)], 0, 0);
    }
    __syncthreads();
    float dik = colk[i];
    float* drow = d + (size_t)i * NN + j0;
#pragma unroll 8
    for (int j = 0; j < 256; ++j) {
      float cand = dik + rowk[j0 + j];
      if (cand < drow[j]) drow[j] = cand;
    }
    __syncthreads();
  }
}

// ---------------------------------------------------------------------------
// 3) Compress mask to 1 byte: bits[2:0]=sp_idx (0..6), bit7 = "-inf" flag
// ---------------------------------------------------------------------------
__global__ __launch_bounds__(256) void mask_code_kernel(
    const float* __restrict__ dist, const unsigned char* __restrict__ binary,
    unsigned char* __restrict__ code) {
  int idx = blockIdx.x * 256 + threadIdx.x;
  float dv = dist[idx];
  int di = (dv > 10.0f) ? -1 : (int)dv;     // inf > 10 -> -1
  int sp = di + 1;
  sp = sp < 0 ? 0 : (sp > 6 ? 6 : sp);
  unsigned char c = (unsigned char)sp;
  if (binary[idx] == 0) c |= 0x80;          // base = -inf
  code[idx] = c;
}

// ---------------------------------------------------------------------------
// 4) WMMA GEMM:  Y[M,N] = X[M,K] @ W[N,K]^T + bias[N], optional ReLU.
//    One wave per 16x64 output tile: A-fragment reused across 4 WMMAs.
// ---------------------------------------------------------------------------
__global__ __launch_bounds__(128) void gemm_xwt_kernel(
    const float* __restrict__ X, const float* __restrict__ W,
    const float* __restrict__ bias, float* __restrict__ Y,
    int M, int N, int K, int relu) {
  int lane = threadIdx.x & 31;
  int wave = blockIdx.x * 4 + (threadIdx.x >> 5);
  int ntiles = N >> 6;              // 64-wide n tiles
  if (wave >= (M >> 4) * ntiles) return;
  int mt = wave / ntiles;
  int nt = wave - mt * ntiles;
  int m0 = mt << 4, n0 = nt << 6;
  v8f acc[4] = {{}, {}, {}, {}};
  for (int k0 = 0; k0 < K; k0 += 32) {
    v16h a = load_frag_rowK(X, K, m0, k0, lane);
    v16h b0 = load_frag_rowK(W, K, n0,      k0, lane);
    v16h b1 = load_frag_rowK(W, K, n0 + 16, k0, lane);
    v16h b2 = load_frag_rowK(W, K, n0 + 32, k0, lane);
    v16h b3 = load_frag_rowK(W, K, n0 + 48, k0, lane);
    acc[0] = wmma_f16(a, b0, acc[0]);
    acc[1] = wmma_f16(a, b1, acc[1]);
    acc[2] = wmma_f16(a, b2, acc[2]);
    acc[3] = wmma_f16(a, b3, acc[3]);
  }
  int mrow = m0 + ((lane >> 4) << 3);
  int nc = lane & 15;
#pragma unroll
  for (int s = 0; s < 4; ++s) {
    int ncol = n0 + s * 16 + nc;
    float bv = bias[ncol];
#pragma unroll
    for (int v = 0; v < 8; ++v) {
      float y = acc[s][v] + bv;
      if (relu) y = fmaxf(y, 0.0f);
      Y[(size_t)(mrow + v) * N + ncol] = y;
    }
  }
}

// ---------------------------------------------------------------------------
// 5) Attention: one wave per (b, h, 16-row q tile). HD=32 -> one WMMA per
//    16x16 score tile. Scores staged in 32KB LDS, softmax via shfl, P@V WMMA
//    with the P fragment reused across both d-halves.
// ---------------------------------------------------------------------------
__global__ __launch_bounds__(32) void attn_kernel(
    const float* __restrict__ qkv, const unsigned char* __restrict__ code,
    const float* __restrict__ emb_spd, float* __restrict__ ctx) {
  __shared__ float sP[16 * NN];              // 32 KB score/prob tile
  __shared__ float sEmb[7 * HH];
  int lane = threadIdx.x;
  int blk = blockIdx.x;                      // b*256 + h*32 + qt
  int qt = blk & 31;
  int h  = (blk >> 5) & 7;
  int b  = blk >> 8;
  if (lane < 7 * HH) sEmb[lane] = emb_spd[lane];
  __syncthreads();

  const float* qbase = qkv + (size_t)b * NN * (3 * EE) + h * HD;
  const float* kbase = qbase + EE;
  const float* vbase = qbase + 2 * EE;
  int q0 = qt * 16;
  int mrow = (lane >> 4) << 3;               // C-layout row base within tile
  int ncol = lane & 15;
  const float scale = 0.17677669529663687f;  // 1/sqrt(32)

  v16h aQ = load_frag_rowK(qbase, 3 * EE, q0, 0, lane);
  for (int kt = 0; kt < NN / 16; ++kt) {
    v16h bK = load_frag_rowK(kbase, 3 * EE, kt * 16, 0, lane);
    v8f acc = {};
    acc = wmma_f16(aQ, bK, acc);
    int key = kt * 16 + ncol;
    const unsigned char* crow =
        code + ((size_t)b * NN + q0 + mrow) * NN + key;
#pragma unroll
    for (int v = 0; v < 8; ++v) {
      unsigned char c = crow[(size_t)v * NN];
      float bias = (c & 0x80) ? -1e30f : sEmb[(c & 7) * HH + h];
      sP[(mrow + v) * NN + key] = acc[v] * scale + bias;
    }
  }
  __syncthreads();

  // softmax over 512 keys; 2 lanes per row
  {
    int r = lane >> 1;
    int half = (lane & 1) << 8;
    float* row = sP + r * NN;
    float m = -__builtin_inff();
    for (int j = 0; j < 256; ++j) m = fmaxf(m, row[half + j]);
    m = fmaxf(m, __shfl_xor(m, 1, 32));
    float l = 0.0f;
    for (int j = 0; j < 256; ++j) {
      float e = __expf(row[half + j] - m);
      row[half + j] = e;
      l += e;
    }
    l += __shfl_xor(l, 1, 32);
    float inv = 1.0f / l;
    for (int j = 0; j < 256; ++j) row[half + j] *= inv;
  }
  __syncthreads();

  // ctx[16,32] = P[16,512] @ V[512,32]; P fragment shared by both halves
  v8f acc0 = {}, acc1 = {};
  for (int kt = 0; kt < NN / 32; ++kt) {
    v16h aP  = load_frag_rowK(sP, NN, 0, kt * 32, lane);
    v16h bV0 = load_frag_colK(vbase, 3 * EE, kt * 32, 0,  lane);
    v16h bV1 = load_frag_colK(vbase, 3 * EE, kt * 32, 16, lane);
    acc0 = wmma_f16(aP, bV0, acc0);
    acc1 = wmma_f16(aP, bV1, acc1);
  }
  float* outp = ctx + ((size_t)b * NN + q0) * EE + h * HD;
#pragma unroll
  for (int v = 0; v < 8; ++v) {
    outp[(size_t)(mrow + v) * EE + ncol]      = acc0[v];
    outp[(size_t)(mrow + v) * EE + 16 + ncol] = acc1[v];
  }
}

// ---------------------------------------------------------------------------
// 6) Residual add + LayerNorm, one row (E=256) per 256-thread block
// ---------------------------------------------------------------------------
__global__ __launch_bounds__(256) void add_ln_kernel(
    const float* __restrict__ A, const float* __restrict__ Bq,
    const float* __restrict__ g, const float* __restrict__ beta,
    float* __restrict__ Y) {
  int row = blockIdx.x;
  int t = threadIdx.x;
  size_t idx = (size_t)row * EE + t;
  float v = A[idx] + Bq[idx];
  __shared__ float sb[8];
  int wid = t >> 5, lane = t & 31;
  float s = wave_reduce_sum(v);
  if (lane == 0) sb[wid] = s;
  __syncthreads();
  float tot = 0.0f;
#pragma unroll
  for (int i = 0; i < 8; ++i) tot += sb[i];
  float mean = tot * (1.0f / EE);
  float d = v - mean;
  __syncthreads();
  float s2 = wave_reduce_sum(d * d);
  if (lane == 0) sb[wid] = s2;
  __syncthreads();
  float tv = 0.0f;
#pragma unroll
  for (int i = 0; i < 8; ++i) tv += sb[i];
  float var = tv * (1.0f / EE);
  Y[idx] = d * rsqrtf(var + 1e-5f) * g[t] + beta[t];
}

// ---------------------------------------------------------------------------
// 7) Attention pooling: one block per batch (512 threads = nodes)
// ---------------------------------------------------------------------------
__global__ __launch_bounds__(512) void pool_kernel(
    const float* __restrict__ x2, const float* __restrict__ wp,
    const float* __restrict__ bp, float* __restrict__ out) {
  int b = blockIdx.x;
  int t = threadIdx.x;                       // node index
  __shared__ float sw[NN];
  __shared__ float red[NN];
  const float* xr = x2 + ((size_t)b * NN + t) * EE;
  float dot = bp[0];
#pragma unroll 4
  for (int e = 0; e < EE; ++e) dot += xr[e] * wp[e];
  float s = tanhf(dot);
  red[t] = s;
  __syncthreads();
  for (int o = 256; o > 0; o >>= 1) {
    if (t < o) red[t] = fmaxf(red[t], red[t + o]);
    __syncthreads();
  }
  float mx = red[0];
  __syncthreads();
  float e = __expf(s - mx);
  red[t] = e;
  __syncthreads();
  for (int o = 256; o > 0; o >>= 1) {
    if (t < o) red[t] += red[t + o];
    __syncthreads();
  }
  float inv = 1.0f / red[0];
  __syncthreads();
  sw[t] = e * inv;
  __syncthreads();
  if (t < EE) {
    float acc = 0.0f;
    const float* xb = x2 + (size_t)b * NN * EE + t;
    for (int n = 0; n < NN; ++n) acc += xb[(size_t)n * EE] * sw[n];
    out[b * EE + t] = acc;
  }
}

// ---------------------------------------------------------------------------
extern "C" void kernel_launch(void* const* d_in, const int* in_sizes, int n_in,
                              void* d_out, int out_size, void* d_ws, size_t ws_size,
                              hipStream_t stream) {
  (void)in_sizes; (void)n_in; (void)out_size; (void)ws_size;
  const float* x      = (const float*)d_in[0];
  const float* adj    = (const float*)d_in[1];
  const float* emb    = (const float*)d_in[2];
  const float* w_qkv  = (const float*)d_in[3];
  const float* b_qkv  = (const float*)d_in[4];
  const float* w_out  = (const float*)d_in[5];
  const float* b_out  = (const float*)d_in[6];
  const float* w_ff1  = (const float*)d_in[7];
  const float* b_ff1  = (const float*)d_in[8];
  const float* w_ff2  = (const float*)d_in[9];
  const float* b_ff2  = (const float*)d_in[10];
  const float* g1     = (const float*)d_in[11];
  const float* be1    = (const float*)d_in[12];
  const float* g2     = (const float*)d_in[13];
  const float* be2    = (const float*)d_in[14];
  const float* w_pool = (const float*)d_in[15];
  const float* b_pool = (const float*)d_in[16];
  float* out = (float*)d_out;

  char* ws = (char*)d_ws;
  size_t off = 0;
  auto wsalloc = [&](size_t bytes) -> void* {
    void* p = ws + off;
    off += (bytes + 255) & ~(size_t)255;
    return p;
  };
  float*         dist   = (float*)wsalloc((size_t)BB * NN * NN * 4);
  unsigned char* binary = (unsigned char*)wsalloc((size_t)BB * NN * NN);
  unsigned char* code   = (unsigned char*)wsalloc((size_t)BB * NN * NN);
  float*         qkvb   = (float*)wsalloc((size_t)BB * NN * 3 * EE * 4);
  float*         ctx    = (float*)wsalloc((size_t)BB * NN * EE * 4);
  float*         aout   = (float*)wsalloc((size_t)BB * NN * EE * 4);
  float*         x1     = (float*)wsalloc((size_t)BB * NN * EE * 4);
  float*         ffb    = (float*)wsalloc((size_t)BB * NN * FF * 4);
  float*         f2o    = (float*)wsalloc((size_t)BB * NN * EE * 4);
  float*         x2     = (float*)wsalloc((size_t)BB * NN * EE * 4);

  const int M = BB * NN;                       // 4096 token rows

  topk_binary_dist_kernel<<<BB * NN, NN, 0, stream>>>(adj, binary, dist);
  floyd_warshall_kernel<<<BB, 1024, 0, stream>>>(dist);
  mask_code_kernel<<<(BB * NN * NN) / 256, 256, 0, stream>>>(dist, binary, code);

  // qkv = x @ w_qkv^T + b_qkv                        [4096 x 768]
  gemm_xwt_kernel<<<((M / 16) * (3 * EE / 64) + 3) / 4, 128, 0, stream>>>(
      x, w_qkv, b_qkv, qkvb, M, 3 * EE, EE, 0);

  attn_kernel<<<BB * HH * (NN / 16), 32, 0, stream>>>(qkvb, code, emb, ctx);

  // attn_out = ctx @ w_out^T + b_out                 [4096 x 256]
  gemm_xwt_kernel<<<((M / 16) * (EE / 64) + 3) / 4, 128, 0, stream>>>(
      ctx, w_out, b_out, aout, M, EE, EE, 0);

  add_ln_kernel<<<M, EE, 0, stream>>>(x, aout, g1, be1, x1);

  // ff = relu(x1 @ w_ff1^T + b_ff1)                  [4096 x 1024]
  gemm_xwt_kernel<<<((M / 16) * (FF / 64) + 3) / 4, 128, 0, stream>>>(
      x1, w_ff1, b_ff1, ffb, M, FF, EE, 1);
  // ff2 = ff @ w_ff2^T + b_ff2                       [4096 x 256]
  gemm_xwt_kernel<<<((M / 16) * (EE / 64) + 3) / 4, 128, 0, stream>>>(
      ffb, w_ff2, b_ff2, f2o, M, EE, FF, 0);

  add_ln_kernel<<<M, EE, 0, stream>>>(x1, f2o, g2, be2, x2);

  pool_kernel<<<BB, NN, 0, stream>>>(x2, w_pool, b_pool, out);
}
